// ExpanderGINNet_81149112091151
// MI455X (gfx1250) — compile-verified, matrix operand order checked by reading, not run on hardware
//
#include <hip/hip_runtime.h>

// ---------------- problem constants (from reference) ----------------
#define N_NODES   100000
#define N_EDGES   1600000
#define HID       128
#define N_CLASSES 10
#define NLAYERS   4
#define N_GRAPHS  128
#define BN_EPS    1e-5f
#define SCAN_CHUNK 1024
#define SCAN_NBLK  ((N_NODES + SCAN_CHUNK - 1) / SCAN_CHUNK)   // 98

typedef __attribute__((ext_vector_type(16))) _Float16 v16h;
typedef __attribute__((ext_vector_type(8)))  _Float16 v8h;
typedef __attribute__((ext_vector_type(8)))  float    v8f;

// ---------------- generic helpers ----------------
__global__ void zero_kernel(float* __restrict__ p, long n) {
    long i = (long)blockIdx.x * blockDim.x + threadIdx.x;
    long stride = (long)gridDim.x * blockDim.x;
    for (; i < n; i += stride) p[i] = 0.0f;
}

__global__ void counts_kernel(const int* __restrict__ gid, float* __restrict__ cnt, int n) {
    int i = blockIdx.x * blockDim.x + threadIdx.x;
    int stride = gridDim.x * blockDim.x;
    for (; i < n; i += stride) atomicAdd(&cnt[gid[i]], 1.0f);
}

// ---------------- pre-mask + f16 conversion of all weights ----------------
// wmT[m][n][k] = (W_m * M_m)[k][n], N-major so B-fragments load contiguous K.
// m=0: emb, m=1..4: W1[i], m=5..8: W2[i].
__global__ void prep_weights_kernel(const float* __restrict__ embW, const float* __restrict__ embM,
                                    const float* __restrict__ W1,   const float* __restrict__ M1,
                                    const float* __restrict__ W2,   const float* __restrict__ M2,
                                    _Float16* __restrict__ wmT) {
    int idx = blockIdx.x * blockDim.x + threadIdx.x;
    if (idx >= 9 * HID * HID) return;
    int m   = idx / (HID * HID);
    int rem = idx % (HID * HID);
    int k   = rem / HID;
    int n   = rem % HID;
    float w;
    if (m == 0) {
        w = embW[k * HID + n] * embM[k * HID + n];
    } else if (m <= 4) {
        int i = m - 1;
        w = W1[(i * HID + k) * HID + n] * M1[(i * HID + k) * HID + n];
    } else {
        int i = m - 5;
        w = W2[(i * HID + k) * HID + n] * M2[(i * HID + k) * HID + n];
    }
    wmT[(m * HID + n) * HID + k] = (_Float16)w;
}

// ================= one-time CSR build (dst-sorted edge index) =================
__global__ void deg_kernel(const int* __restrict__ dst, unsigned* __restrict__ deg, int ne) {
    int i = blockIdx.x * blockDim.x + threadIdx.x;
    int stride = gridDim.x * blockDim.x;
    for (; i < ne; i += stride) atomicAdd(&deg[dst[i]], 1u);
}

// pass 1: per-1024-chunk sums
__global__ void scan_partial_kernel(const unsigned* __restrict__ deg,
                                    unsigned* __restrict__ blockSums, int n) {
    __shared__ unsigned ls[256];
    int b = blockIdx.x, t = threadIdx.x;
    int base = b * SCAN_CHUNK + t * 4;
    unsigned s = 0;
#pragma unroll
    for (int j = 0; j < 4; ++j) { int i = base + j; if (i < n) s += deg[i]; }
    ls[t] = s;
    __syncthreads();
    for (int off = 128; off > 0; off >>= 1) {
        if (t < off) ls[t] += ls[t + off];
        __syncthreads();
    }
    if (t == 0) blockSums[b] = ls[0];
}

// pass 2: exclusive scan of the 98 chunk sums (exact integer, one lane)
__global__ void scan_block_kernel(const unsigned* __restrict__ blockSums,
                                  unsigned* __restrict__ blockOffs, int nblk) {
    if (threadIdx.x == 0 && blockIdx.x == 0) {
        unsigned acc = 0;
        for (int i = 0; i < nblk; ++i) { blockOffs[i] = acc; acc += blockSums[i]; }
    }
}

// pass 3: in-chunk exclusive scan (LDS Hillis-Steele over 256 thread sums)
__global__ void scan_final_kernel(const unsigned* __restrict__ deg,
                                  const unsigned* __restrict__ blockOffs,
                                  unsigned* __restrict__ nodeOff, int n) {
    __shared__ unsigned ls[256];
    int b = blockIdx.x, t = threadIdx.x;
    int base = b * SCAN_CHUNK + t * 4;
    unsigned v[4];
    unsigned s = 0;
#pragma unroll
    for (int j = 0; j < 4; ++j) { int i = base + j; v[j] = (i < n) ? deg[i] : 0u; s += v[j]; }
    ls[t] = s;
    __syncthreads();
    for (int off = 1; off < 256; off <<= 1) {
        unsigned add = (t >= off) ? ls[t - off] : 0u;
        __syncthreads();
        ls[t] += add;
        __syncthreads();
    }
    unsigned excl = ls[t] - s + blockOffs[b];
#pragma unroll
    for (int j = 0; j < 4; ++j) {
        int i = base + j;
        if (i < n) { nodeOff[i] = excl; excl += v[j]; }
    }
}

__global__ void csr_fill_kernel(const int* __restrict__ src, const int* __restrict__ dst,
                                const unsigned* __restrict__ nodeOff, unsigned* __restrict__ cursor,
                                int* __restrict__ csrSrc, int ne) {
    int e = blockIdx.x * blockDim.x + threadIdx.x;
    int stride = gridDim.x * blockDim.x;
    for (; e < ne; e += stride) {
        int d = dst[e];
        unsigned pos = nodeOff[d] + atomicAdd(&cursor[d], 1u);
        csrSrc[pos] = src[e];
    }
}

// ---------------- atomic-free GIN aggregation: out = h + sum_{src->n} h[src] ----------------
// One wave per node; lane holds 4 consecutive features, register accumulation.
// h (51 MB) + csrSrc (6.4 MB) are L2-resident; no float atomics at all.
__global__ __launch_bounds__(256)
void gather_kernel(const float* __restrict__ h, const int* __restrict__ csrSrc,
                   const unsigned* __restrict__ nodeOff, const unsigned* __restrict__ deg,
                   float* __restrict__ out, int nnodes) {
    int t = blockIdx.x * blockDim.x + threadIdx.x;
    int node = t >> 5;
    int lane = t & 31;
    if (node >= nnodes) return;
    float4 acc = *(const float4*)(h + (size_t)node * HID + lane * 4);  // (1+eps)*h, eps=0
    unsigned beg = nodeOff[node];
    unsigned end = beg + deg[node];
    for (unsigned j = beg; j < end; ++j) {       // node-uniform trip count
        int s = csrSrc[j];                       // broadcast load
        const float4 v = *(const float4*)(h + (size_t)s * HID + lane * 4);
        acc.x += v.x; acc.y += v.y; acc.z += v.z; acc.w += v.w;
    }
    *(float4*)(out + (size_t)node * HID + lane * 4) = acc;
}

// ---------------- WMMA GEMM: out[N_NODES x 128] = f(A) @ W[128x128] + bias ----------------
// MODE_RAW:    f(A) = A0
// MODE_BNRELU: f(A) = relu(A0*scale + shift), scale=A1[0..127], shift=A1[128..255]
// One wave = 16-row x 128-col output tile via 8x4 = 32 v_wmma_f32_16x16x32_f16.
enum { MODE_RAW = 0, MODE_BNRELU = 2 };

template <int MODE, bool POOL>
__global__ __launch_bounds__(128)
void gemm_wmma_kernel(const float* __restrict__ A0,
                      const float* __restrict__ A1,       // bnsc for MODE_BNRELU
                      const _Float16* __restrict__ wmT,   // [n][k] N-major masked weights
                      const float* __restrict__ bias,     // [128]
                      float* __restrict__ out,
                      const int* __restrict__ gid,        // used if POOL
                      float* __restrict__ pool,           // used if POOL: [N_GRAPHS][128]
                      int ntiles) {
    int lane = threadIdx.x & 31;
    int wave = threadIdx.x >> 5;
    int tile = blockIdx.x * (blockDim.x >> 5) + wave;
    if (tile >= ntiles) return;                 // wave-uniform: EXEC stays full

    int r  = lane & 15;                         // A: row-in-tile; B/D: column-in-tile
    int hi = lane >> 4;
    int row = tile * 16 + r;
    const float* arow = A0 + (size_t)row * HID;

    // A fragments: lane(l%16)=M, element e -> K = (e/8)*16 + (l/16)*8 + (e%8)
    v16h a[4];
#pragma unroll
    for (int kc = 0; kc < 4; ++kc) {
        int k0 = kc * 32 + hi * 8;
#pragma unroll
        for (int j = 0; j < 8; ++j) {
            int kA = k0 + j;
            int kB = k0 + 16 + j;
            float va = arow[kA];
            float vb = arow[kB];
            if (MODE == MODE_BNRELU) {
                va = fmaxf(va * A1[kA] + A1[HID + kA], 0.0f);
                vb = fmaxf(vb * A1[kB] + A1[HID + kB], 0.0f);
            }
            a[kc][j]     = (_Float16)va;
            a[kc][8 + j] = (_Float16)vb;
        }
    }

#pragma unroll
    for (int nt = 0; nt < 8; ++nt) {
        const _Float16* bp = wmT + (size_t)(nt * 16 + r) * HID;
        v8f acc = {};
#pragma unroll
        for (int kc = 0; kc < 4; ++kc) {
            int k0 = kc * 32 + hi * 8;
            v8h b0 = *(const v8h*)(bp + k0);
            v8h b1 = *(const v8h*)(bp + k0 + 16);
            v16h bv;
#pragma unroll
            for (int j = 0; j < 8; ++j) { bv[j] = b0[j]; bv[8 + j] = b1[j]; }
            acc = __builtin_amdgcn_wmma_f32_16x16x32_f16(
                false, a[kc], false, bv, (short)0, acc, false, false);
        }
        int ncol = nt * 16 + r;
        float bc = bias[ncol];
#pragma unroll
        for (int rr = 0; rr < 8; ++rr) {
            int orow = tile * 16 + rr + 8 * hi;   // D: VGPR rr -> M = rr + 8*(l/16)
            float val = acc[rr] + bc;
            out[(size_t)orow * HID + ncol] = val;
            if (POOL) atomicAdd(&pool[gid[orow] * HID + ncol], val);
        }
    }
}

// ---------------- BN statistics: per-column sum & sumsq ----------------
template <bool SN>
__global__ void stats_kernel(const float* __restrict__ x, const float* __restrict__ snorm,
                             float* __restrict__ stats, int nrows) {
    int c = threadIdx.x;  // 128 threads: one column each, coalesced row reads
    float s = 0.0f, q = 0.0f;
    for (int n = blockIdx.x; n < nrows; n += gridDim.x) {
        float v = x[(size_t)n * HID + c];
        if (SN) v *= snorm[n];
        s += v;
        q += v * v;
    }
    atomicAdd(&stats[c], s);
    atomicAdd(&stats[HID + c], q);
}

__global__ void bn_finalize_kernel(const float* __restrict__ stats,
                                   const float* __restrict__ g, const float* __restrict__ b,
                                   float* __restrict__ bnsc, float invN) {
    int c = threadIdx.x;  // 128 threads
    float mean = stats[c] * invN;
    float var  = stats[HID + c] * invN - mean * mean;
    float sc   = g[c] * rsqrtf(var + BN_EPS);
    bnsc[c]       = sc;
    bnsc[HID + c] = b[c] - mean * sc;
}

// ---------------- graph_norm + node-BN + ReLU + residual + pooling ----------------
__global__ void final_kernel(float* __restrict__ h, const float* __restrict__ x2,
                             const float* __restrict__ snorm, const float* __restrict__ bnsc,
                             const int* __restrict__ gid, float* __restrict__ pool, long nh) {
    long i = (long)blockIdx.x * blockDim.x + threadIdx.x;
    long stride = (long)gridDim.x * blockDim.x;
    for (; i < nh; i += stride) {
        int n = (int)(i >> 7);
        int c = (int)(i & 127);
        float y = x2[i] * snorm[n];
        y = fmaxf(y * bnsc[c] + bnsc[HID + c], 0.0f);
        float hv = h[i] + y;
        h[i] = hv;
        atomicAdd(&pool[gid[n] * HID + c], hv);
    }
}

// ---------------- readout ----------------
__global__ void readout_kernel(const float* __restrict__ pool, const float* __restrict__ cnt,
                               const float* __restrict__ roW, const float* __restrict__ roB,
                               float* __restrict__ out) {
    int idx = blockIdx.x * blockDim.x + threadIdx.x;
    if (idx >= N_GRAPHS * N_CLASSES) return;
    int g = idx / N_CLASSES;
    int c = idx % N_CLASSES;
    float inv = 1.0f / fmaxf(cnt[g], 1.0f);
    float acc = 0.0f;
    for (int i = 0; i <= NLAYERS; ++i) {
        const float* pg = pool + (size_t)i * N_GRAPHS * HID + (size_t)g * HID;
        float s = 0.0f;
        for (int k = 0; k < HID; ++k) s += pg[k] * roW[((size_t)i * HID + k) * N_CLASSES + c];
        acc += s * inv + roB[i * N_CLASSES + c];
    }
    out[idx] = acc;
}

// ---------------- driver ----------------
extern "C" void kernel_launch(void* const* d_in, const int* in_sizes, int n_in,
                              void* d_out, int out_size, void* d_ws, size_t ws_size,
                              hipStream_t stream) {
    const int*   src   = (const int*)d_in[0];
    const int*   dst   = (const int*)d_in[1];
    const float* h0    = (const float*)d_in[2];
    const float* snorm = (const float*)d_in[3];
    const int*   gid   = (const int*)d_in[4];
    const float* embW  = (const float*)d_in[5];
    const float* embB  = (const float*)d_in[6];
    const float* embM  = (const float*)d_in[7];
    const float* W1    = (const float*)d_in[8];
    const float* b1    = (const float*)d_in[9];
    const float* M1    = (const float*)d_in[10];
    const float* W2    = (const float*)d_in[11];
    const float* b2    = (const float*)d_in[12];
    const float* M2    = (const float*)d_in[13];
    const float* bnmg  = (const float*)d_in[14];
    const float* bnmb  = (const float*)d_in[15];
    const float* bnng  = (const float*)d_in[16];
    const float* bnnb  = (const float*)d_in[17];
    const float* roW   = (const float*)d_in[18];
    const float* roB   = (const float*)d_in[19];

    const long NH = (long)N_NODES * HID;        // 12.8M floats
    float* ws    = (float*)d_ws;
    float* dH    = ws;                          // current node features
    float* dT1   = ws + NH;                     // aggregated (h+neigh) / x2
    float* dT2   = ws + 2 * NH;                 // x1
    float* dPool = ws + 3 * NH;                 // (L+1) x 128 x 128
    float* dCnt  = dPool + (NLAYERS + 1) * N_GRAPHS * HID;
    float* dStat = dCnt + N_GRAPHS;             // 256: sum | sumsq
    float* dBnsc = dStat + 256;                 // 256: scale | shift
    _Float16* dWmT = (_Float16*)(dBnsc + 256);  // 9 x 128 x 128 f16, N-major
    // integer region (after 73728 float-equivalents of f16 weights)
    unsigned* dDeg     = (unsigned*)(dBnsc + 256 + 9 * HID * HID / 2);
    unsigned* dCursor  = dDeg + N_NODES;
    unsigned* dNodeOff = dCursor + N_NODES;
    unsigned* dBSums   = dNodeOff + N_NODES;
    unsigned* dBOffs   = dBSums + 128;
    int*      dCsrSrc  = (int*)(dBOffs + 128);

    const int NTILES = N_NODES / 16;            // 6250 (exact)
    const int GEMM_GRID = (NTILES + 3) / 4;     // 4 waves / 128-thread block

    // zero: pooled sums + counts, and deg + cursor (int zeros == float zeros)
    zero_kernel<<<64, 256, 0, stream>>>(dPool, (long)(NLAYERS + 1) * N_GRAPHS * HID + N_GRAPHS);
    zero_kernel<<<64, 256, 0, stream>>>((float*)dDeg, 2L * N_NODES);
    counts_kernel<<<256, 256, 0, stream>>>(gid, dCnt, N_NODES);

    // masked weights -> f16, N-major
    prep_weights_kernel<<<(9 * HID * HID + 255) / 256, 256, 0, stream>>>(
        embW, embM, W1, M1, W2, M2, dWmT);

    // ---- one-time CSR build (reused by all 4 layers) ----
    deg_kernel<<<512, 256, 0, stream>>>(dst, dDeg, N_EDGES);
    scan_partial_kernel<<<SCAN_NBLK, 256, 0, stream>>>(dDeg, dBSums, N_NODES);
    scan_block_kernel<<<1, 32, 0, stream>>>(dBSums, dBOffs, SCAN_NBLK);
    scan_final_kernel<<<SCAN_NBLK, 256, 0, stream>>>(dDeg, dBOffs, dNodeOff, N_NODES);
    csr_fill_kernel<<<1024, 256, 0, stream>>>(src, dst, dNodeOff, dCursor, dCsrSrc, N_EDGES);

    // embedding GEMM (+ pool hidden_rep[0])
    gemm_wmma_kernel<MODE_RAW, true><<<GEMM_GRID, 128, 0, stream>>>(
        h0, nullptr, dWmT, embB, dH, gid, dPool, NTILES);

    for (int i = 0; i < NLAYERS; ++i) {
        // x_in = h + scatter_sum(h[src] -> dst), atomic-free gather via CSR
        gather_kernel<<<(N_NODES * 32 + 255) / 256, 256, 0, stream>>>(
            dH, dCsrSrc, dNodeOff, dDeg, dT1, N_NODES);

        // x1 = x_in @ Wm1[i] + b1[i]
        gemm_wmma_kernel<MODE_RAW, false><<<GEMM_GRID, 128, 0, stream>>>(
            dT1, nullptr, dWmT + (size_t)(1 + i) * HID * HID, b1 + i * HID,
            dT2, nullptr, nullptr, NTILES);

        // BN(mlp) stats -> scale/shift
        zero_kernel<<<1, 256, 0, stream>>>(dStat, 256);
        stats_kernel<false><<<1024, 128, 0, stream>>>(dT2, nullptr, dStat, N_NODES);
        bn_finalize_kernel<<<1, 128, 0, stream>>>(dStat, bnmg + i * HID, bnmb + i * HID,
                                                  dBnsc, 1.0f / (float)N_NODES);

        // x2 = relu(bn(x1)) @ Wm2[i] + b2[i]   (BN+ReLU fused into A-load)
        gemm_wmma_kernel<MODE_BNRELU, false><<<GEMM_GRID, 128, 0, stream>>>(
            dT2, dBnsc, dWmT + (size_t)(5 + i) * HID * HID, b2 + i * HID,
            dT1, nullptr, nullptr, NTILES);

        // BN(node) stats over x2 * snorm
        zero_kernel<<<1, 256, 0, stream>>>(dStat, 256);
        stats_kernel<true><<<1024, 128, 0, stream>>>(dT1, snorm, dStat, N_NODES);
        bn_finalize_kernel<<<1, 128, 0, stream>>>(dStat, bnng + i * HID, bnnb + i * HID,
                                                  dBnsc, 1.0f / (float)N_NODES);

        // h += relu(bn(x2*snorm)); pool hidden_rep[i+1]
        final_kernel<<<2048, 256, 0, stream>>>(dH, dT1, snorm, dBnsc, gid,
                                               dPool + (size_t)(i + 1) * N_GRAPHS * HID, NH);
    }

    readout_kernel<<<(N_GRAPHS * N_CLASSES + 255) / 256, 256, 0, stream>>>(
        dPool, dCnt, roW, roB, (float*)d_out);
}
